// Rnn_44598940402038
// MI455X (gfx1250) — matrix operand order repaired and checked by reading, not compiled, not run on hardware
//
#include <hip/hip_runtime.h>
#include <hip/hip_bf16.h>

typedef __bf16 bf16_t;
typedef __attribute__((ext_vector_type(16))) __bf16        v16bf;
typedef __attribute__((ext_vector_type(8)))  float         v8f;
typedef __attribute__((ext_vector_type(4)))  unsigned int  u32x4;

#define VOCAB   4411
#define EMB     64
#define HID     128
#define T_LEN   512
#define BATCH   1024
#define OUT_DIM 2

#define W_STRIDE 136   // bf16 elems per W_hh row in LDS (128 + 8 pad -> 272B, bank stride 4)
#define H_STRIDE 132   // f32 elems per h row in LDS    (128 + 4 pad, bank stride 4)

// Hardware tanh (gfx1250 TRANS op, co-executes with the WMMA pipe).
__device__ __forceinline__ float hw_tanh(float x) {
#if __has_builtin(__builtin_amdgcn_tanhf)
  return __builtin_amdgcn_tanhf(x);
#else
  float r;
  // v_nop covers the 1-cycle TRANS result hazard since the compiler can't
  // track deps through the asm.
  asm volatile("v_tanh_f32 %0, %1\n\tv_nop" : "=v"(r) : "v"(x));
  return r;
#endif
}

// ---------------------------------------------------------------------------
// Kernel 1: fold embedding + input projection + both biases into a per-token
// table:  proj[v][h] = sum_e emb[v][e] * W_ih[h][e] + b_ih[h] + b_hh[h]
// (72 MFLOP, runs once; output 2.25 MB stays resident in the 192 MB L2)
// ---------------------------------------------------------------------------
__global__ void proj_kernel(const float* __restrict__ emb,
                            const float* __restrict__ W_ih,
                            const float* __restrict__ b_ih,
                            const float* __restrict__ b_hh,
                            float* __restrict__ proj) {
  int idx = blockIdx.x * blockDim.x + threadIdx.x;
  if (idx >= VOCAB * HID) return;
  int v = idx / HID, h = idx - v * HID;
  const float* e = emb  + v * EMB;
  const float* w = W_ih + h * EMB;
  float s = b_ih[h] + b_hh[h];
#pragma unroll
  for (int k = 0; k < EMB; ++k) s += e[k] * w[k];
  proj[idx] = s;
}

// ---------------------------------------------------------------------------
// Kernel 2: the recurrent scan. One wave owns a 16-row batch tile for all 512
// steps (no inter-wave sync ever).  Per step:
//   acc[j](16x16 f32, C layout)  <- gather of proj_table rows  (x_t)
//   acc[j] += h(16x128,bf16) x W_hh^T(128x16,bf16)  via 32 v_wmma 16x16x32
//   h <- v_tanh_f32(acc); staged f32 in LDS for the C->A layout reshuffle
// ---------------------------------------------------------------------------
__launch_bounds__(64)
__global__ void rnn_kernel(const int*   __restrict__ tokens,
                           const float* __restrict__ proj,
                           const float* __restrict__ W_hh,
                           const float* __restrict__ W_lin,
                           const float* __restrict__ b_lin,
                           float* __restrict__ out) {
  __shared__ bf16_t w_lds[HID * W_STRIDE];       // W_hh, bf16, row-major, padded
  __shared__ float  h_lds[2][16 * H_STRIDE];     // per-wave h state (f32)

  const int tid = threadIdx.x;

  // stage W_hh into LDS as bf16 (B operand is B[k][n] = W_hh[n][k]; we keep
  // W_hh row-major so a column-n / K-contiguous fragment read is contiguous)
  for (int i = tid; i < HID * HID; i += 64) {
    int r = i >> 7, c = i & 127;
    w_lds[r * W_STRIDE + c] = (bf16_t)W_hh[i];
  }
  __syncthreads();

  const int wave = tid >> 5;
  const int lane = tid & 31;
  const int c16  = lane & 15;
  const int hi   = lane >> 4;                    // 0: lanes 0-15, 1: lanes 16-31
  const int b0   = (blockIdx.x * 2 + wave) * 16; // first batch row of this tile
  float* hbuf = h_lds[wave];

  v8f acc[8];                                    // 8 N-tiles of the 16x128 state

  for (int t = 0; t < T_LEN; ++t) {
    // tokens for the 8 M-rows this lane covers (C layout: M = r + 8*hi)
    int tok[8];
#pragma unroll
    for (int r = 0; r < 8; ++r)
      tok[r] = tokens[(b0 + r + 8 * hi) * T_LEN + t];

    // init accumulators with x_t = proj_table[tok] gathered in C layout
#pragma unroll
    for (int j = 0; j < 8; ++j) {
      v8f a;
#pragma unroll
      for (int r = 0; r < 8; ++r)
        a[r] = proj[tok[r] * HID + j * 16 + c16];
      acc[j] = a;
    }

    if (t > 0) {
#pragma unroll
      for (int k = 0; k < 4; ++k) {              // K = 128 in 4 chunks of 32
        // A fragment: lane holds row c16, K in {s0..s0+15}
        const int    s0   = 32 * k + 16 * hi;
        const float* hrow = hbuf + c16 * H_STRIDE;
        v16bf afrag;
#pragma unroll
        for (int p = 0; p < 8; ++p) afrag[p]     = (bf16_t)hrow[s0 + p];
#pragma unroll
        for (int p = 0; p < 8; ++p) afrag[p + 8] = (bf16_t)hrow[s0 + 8 + p];

#pragma unroll
        for (int j = 0; j < 8; ++j) {
          // B fragment: lane holds column j*16+c16, 16 consecutive K (+16*hi)
          const bf16_t* wp = w_lds + (j * 16 + c16) * W_STRIDE + 32 * k + 16 * hi;
          union { u32x4 u[2]; v16bf v; } bf;
          bf.u[0] = *(const u32x4*)(wp);         // 16B-aligned ds_load_b128
          bf.u[1] = *(const u32x4*)(wp + 8);
          acc[j] = __builtin_amdgcn_wmma_f32_16x16x32_bf16(
              false, afrag, false, bf.v, (short)0, acc[j], false, false);
        }
      }
    }

    // tanh and scatter h (C layout) back to LDS for next step's A fragments
#pragma unroll
    for (int j = 0; j < 8; ++j) {
#pragma unroll
      for (int r = 0; r < 8; ++r) {
        float hv = hw_tanh(acc[j][r]);
        hbuf[(r + 8 * hi) * H_STRIDE + j * 16 + c16] = hv;
      }
    }
    asm volatile("s_wait_dscnt 0" ::: "memory"); // wave-local RAW on hbuf
  }

  // head: out[b][o] = h_final[b] . W_lin[o] + b_lin[o]   (32 dots per wave)
  {
    const int m = lane & 15;
    const int o = lane >> 4;
    const float* hrow = hbuf + m * H_STRIDE;
    const float* w    = W_lin + o * HID;
    float s = b_lin[o];
#pragma unroll 16
    for (int k = 0; k < HID; ++k) s += hrow[k] * w[k];
    out[(b0 + m) * OUT_DIM + o] = s;
  }
}

// ---------------------------------------------------------------------------
extern "C" void kernel_launch(void* const* d_in, const int* in_sizes, int n_in,
                              void* d_out, int out_size, void* d_ws, size_t ws_size,
                              hipStream_t stream) {
  const int*   inputs = (const int*)  d_in[0];
  const float* embt   = (const float*)d_in[1];
  const float* W_ih   = (const float*)d_in[2];
  const float* W_hh   = (const float*)d_in[3];
  const float* b_ih   = (const float*)d_in[4];
  const float* b_hh   = (const float*)d_in[5];
  const float* W_lin  = (const float*)d_in[6];
  const float* b_lin  = (const float*)d_in[7];
  float*       outp   = (float*)d_out;
  float*       projt  = (float*)d_ws;            // 4411*128*4 = 2,258,432 B

  proj_kernel<<<(VOCAB * HID + 255) / 256, 256, 0, stream>>>(
      embt, W_ih, b_ih, b_hh, projt);

  // 64 independent batch tiles of 16 rows -> 32 blocks x 2 waves
  rnn_kernel<<<32, 64, 0, stream>>>(inputs, projt, W_hh, W_lin, b_lin, outp);
}
